// MHSA_1537598292331
// MI455X (gfx1250) — matrix operand order repaired and checked by reading, not compiled, not run on hardware
//
#include <hip/hip_runtime.h>
#include <cstdint>

// MHSA forward for B=2, S=2048, D=1024, H=16 (DH=64) on gfx1250 (MI455X).
// bf16 WMMA (v_wmma_f32_16x16x32_bf16) for all four matmul stages, fp32
// softmax with flash-style online stats, attn_weights written exactly once.
// Out-projection stages its A tile with global_load_async_to_lds_b128
// (ASYNCcnt) to exercise the CDNA5 async global->LDS path.

#define BB 2
#define SS 2048
#define DD 1024
#define HH 16
#define DH 64

typedef __bf16 bf16;
typedef __attribute__((ext_vector_type(16))) __bf16 v16bf;
typedef __attribute__((ext_vector_type(8)))  float  v8f;

union FragB16 { v16bf v; unsigned short u[16]; uint4 q[2]; };
union FragF32 { v8f v; float f[8]; };

__device__ __forceinline__ unsigned short f2bf(float f) {
  union { __bf16 h; unsigned short u; } cv;
  cv.h = (__bf16)f;                 // native cvt (RNE) on gfx1250
  return cv.u;
}

// ---------------------------------------------------------------------------
// Kernel 1: QKV projection.  C[4096,1024] = X @ W + bias.
// grid.z selects (Q,K,V).  Q,K stored bf16 as [B,H,S,DH]; V stored transposed
// [B,H,DH,S] so attention context B-fragments are contiguous loads.
// Block = 256 threads = 8 waves; 64x64 tile; each wave does 16(m) x 32(n).
// ---------------------------------------------------------------------------
__global__ __launch_bounds__(256) void qkv_proj_kernel(
    const float* __restrict__ query, const float* __restrict__ key,
    const float* __restrict__ value,
    const float* __restrict__ Wq, const float* __restrict__ bq,
    const float* __restrict__ Wk, const float* __restrict__ bk,
    const float* __restrict__ Wv, const float* __restrict__ bv,
    unsigned short* __restrict__ Qp, unsigned short* __restrict__ Kp,
    unsigned short* __restrict__ Vt)
{
  const int z = blockIdx.z;
  const float* A    = (z == 0) ? query : (z == 1) ? key : value;
  const float* W    = (z == 0) ? Wq    : (z == 1) ? Wk  : Wv;
  const float* bias = (z == 0) ? bq    : (z == 1) ? bk  : bv;

  __shared__ unsigned short ldsA[64][32];   // [m][k] bf16
  __shared__ unsigned short ldsW[64][32];   // [n][k] bf16 (transposed on fill)

  const int tid  = threadIdx.x;
  const int lane = tid & 31;
  const int warp = tid >> 5;
  const int m0 = blockIdx.y * 64;
  const int n0 = blockIdx.x * 64;
  const int m0w = (warp >> 1) * 16;   // 0,16,32,48
  const int n0w = (warp & 1) * 32;    // 0,32
  const int lr   = lane & 15;
  const int koff = (lane >> 4) * 8;   // unused (A and B frags both use kh16)
  const int kh16 = (lane >> 4) * 16;  // B-frag K striping

  FragF32 c0, c1;
  #pragma unroll
  for (int i = 0; i < 8; i++) { c0.f[i] = 0.f; c1.f[i] = 0.f; }

  for (int kk = 0; kk < DD; kk += 32) {
    { // stage A tile 64x32 fp32 -> bf16
      int row = tid >> 2, col = (tid & 3) * 8;
      const float* src = A + (size_t)(m0 + row) * DD + kk + col;
      #pragma unroll
      for (int j = 0; j < 8; j++) ldsA[row][col + j] = f2bf(src[j]);
    }
    { // stage W tile 32x64 fp32 -> bf16, transposed to [n][k]
      int row = tid >> 3, col = (tid & 7) * 8;
      const float* src = W + (size_t)(kk + row) * DD + n0 + col;
      #pragma unroll
      for (int j = 0; j < 8; j++) ldsW[col + j][row] = f2bf(src[j]);
    }
    __syncthreads();

    FragB16 a, b0, b1;
    a.q[0]  = *(const uint4*)&ldsA[m0w + lr][kh16];
    a.q[1]  = *(const uint4*)&ldsA[m0w + lr][kh16 + 8];
    b0.q[0] = *(const uint4*)&ldsW[n0w + lr][kh16];
    b0.q[1] = *(const uint4*)&ldsW[n0w + lr][kh16 + 8];
    b1.q[0] = *(const uint4*)&ldsW[n0w + 16 + lr][kh16];
    b1.q[1] = *(const uint4*)&ldsW[n0w + 16 + lr][kh16 + 8];

    c0.v = __builtin_amdgcn_wmma_f32_16x16x32_bf16(false, a.v, false, b0.v,
                                                   (short)0, c0.v, false, false);
    c1.v = __builtin_amdgcn_wmma_f32_16x16x32_bf16(false, a.v, false, b1.v,
                                                   (short)0, c1.v, false, false);
    __syncthreads();
  }

  // store: C layout -> lane l holds rows (r + 8*(l>>4)), col = lr
  const int rowhalf = (lane >> 4) * 8;
  unsigned short* qkdst = (z == 0) ? Qp : Kp;
  #pragma unroll
  for (int cn = 0; cn < 2; cn++) {
    FragF32& c = cn ? c1 : c0;
    int gn = n0 + n0w + cn * 16 + lr;
    float bia = bias[gn];
    int h = gn >> 6, dh = gn & 63;
    #pragma unroll
    for (int r = 0; r < 8; r++) {
      int gm = m0 + m0w + rowhalf + r;
      int bb = gm >> 11, s = gm & 2047;
      unsigned short bfv = f2bf(c.f[r] + bia);
      if (z < 2) qkdst[(((size_t)(bb * HH + h)) * SS + s) * DH + dh] = bfv;
      else       Vt[(((size_t)(bb * HH + h)) * DH + dh) * SS + s]   = bfv;
    }
  }
}

// ---------------------------------------------------------------------------
// Kernel 2: flash attention.  One wave per (b,h,16-row q tile).
// Pass 1: online softmax stats + context accumulation (nothing to HBM).
// Pass 2: recompute scores (K tiles L2-resident), write normalized weights.
// Block = 128 threads = 4 waves.
// ---------------------------------------------------------------------------
__global__ __launch_bounds__(128) void attn_kernel(
    const unsigned short* __restrict__ Qp, const unsigned short* __restrict__ Kp,
    const unsigned short* __restrict__ Vt, const float* __restrict__ mask,
    float* __restrict__ attn, unsigned short* __restrict__ ctx)
{
  __shared__ unsigned short ldsP[4][16][32];   // per-wave P transpose buffer

  const int lane = threadIdx.x & 31;
  const int warp = threadIdx.x >> 5;
  const int wid  = blockIdx.x * 4 + warp;      // 0..4095
  const int bh = wid >> 7;                     // b*16+h
  const int q0 = (wid & 127) * 16;
  const int b  = bh >> 4;
  const int h  = bh & 15;
  const int lr   = lane & 15;
  const int koff = (lane >> 4) * 8;
  const int kh16 = (lane >> 4) * 16;
  const int rowhalf = (lane >> 4) * 8;
  const float scale = 0.125f;                  // 1/sqrt(64)

  // Q fragments (A-layout), rows q0..q0+15, d split into two K=32 steps
  FragB16 aq0, aq1;
  {
    const unsigned short* qrow = Qp + ((size_t)bh * SS + q0 + lr) * DH;
    aq0.q[0] = *(const uint4*)(qrow + kh16);
    aq0.q[1] = *(const uint4*)(qrow + kh16 + 8);
    aq1.q[0] = *(const uint4*)(qrow + 32 + kh16);
    aq1.q[1] = *(const uint4*)(qrow + 32 + kh16 + 8);
  }

  float mrun[8], lrun[8];
  FragF32 acc[4];
  #pragma unroll
  for (int r = 0; r < 8; r++) { mrun[r] = -3.0e38f; lrun[r] = 0.f; }
  #pragma unroll
  for (int d = 0; d < 4; d++)
    #pragma unroll
    for (int r = 0; r < 8; r++) acc[d].f[r] = 0.f;

  const unsigned short* Kbase = Kp + (size_t)bh * SS * DH;
  const unsigned short* Vbase = Vt + (size_t)bh * DH * SS;
  const float* mk = mask + (size_t)b * SS;
  const v8f zero8 = {0.f, 0.f, 0.f, 0.f, 0.f, 0.f, 0.f, 0.f};

  // -------------------- pass 1: stats + context --------------------
  for (int kt = 0; kt < 64; kt++) {
    const int k0 = kt * 32;
    FragB16 bk[2][2];
    #pragma unroll
    for (int kc = 0; kc < 2; kc++) {
      const unsigned short* krow = Kbase + (size_t)(k0 + kc * 16 + lr) * DH;
      #pragma unroll
      for (int ds = 0; ds < 2; ds++) {
        bk[kc][ds].q[0] = *(const uint4*)(krow + ds * 32 + kh16);
        bk[kc][ds].q[1] = *(const uint4*)(krow + ds * 32 + kh16 + 8);
      }
    }
    FragF32 c0, c1;
    c0.v = __builtin_amdgcn_wmma_f32_16x16x32_bf16(false, aq0.v, false, bk[0][0].v,
                                                   (short)0, zero8, false, false);
    c0.v = __builtin_amdgcn_wmma_f32_16x16x32_bf16(false, aq1.v, false, bk[0][1].v,
                                                   (short)0, c0.v, false, false);
    c1.v = __builtin_amdgcn_wmma_f32_16x16x32_bf16(false, aq0.v, false, bk[1][0].v,
                                                   (short)0, zero8, false, false);
    c1.v = __builtin_amdgcn_wmma_f32_16x16x32_bf16(false, aq1.v, false, bk[1][1].v,
                                                   (short)0, c1.v, false, false);

    const float mv0 = mk[k0 + lr] * -1.0e9f;
    const float mv1 = mk[k0 + 16 + lr] * -1.0e9f;

    float p0[8], p1[8];
    #pragma unroll
    for (int r = 0; r < 8; r++) {
      float s0 = c0.f[r] * scale + mv0;
      float s1 = c1.f[r] * scale + mv1;
      float t = fmaxf(s0, s1);
      #pragma unroll
      for (int m = 1; m <= 8; m <<= 1) t = fmaxf(t, __shfl_xor(t, m, 32));
      float mnew = fmaxf(mrun[r], t);
      float corr = __expf(mrun[r] - mnew);
      lrun[r] *= corr;
      acc[0].f[r] *= corr; acc[1].f[r] *= corr;
      acc[2].f[r] *= corr; acc[3].f[r] *= corr;
      p0[r] = __expf(s0 - mnew);
      p1[r] = __expf(s1 - mnew);
      float u = p0[r] + p1[r];
      #pragma unroll
      for (int m = 1; m <= 8; m <<= 1) u += __shfl_xor(u, m, 32);
      lrun[r] += u;
      mrun[r] = mnew;
    }

    // transpose P (C-layout) -> A-layout through LDS, bf16
    #pragma unroll
    for (int r = 0; r < 8; r++) {
      int row = r + rowhalf;
      ldsP[warp][row][lr]      = f2bf(p0[r]);
      ldsP[warp][row][lr + 16] = f2bf(p1[r]);
    }
    __syncthreads();
    FragB16 ap;
    ap.q[0] = *(const uint4*)&ldsP[warp][lr][kh16];
    ap.q[1] = *(const uint4*)&ldsP[warp][lr][kh16 + 8];
    __syncthreads();

    // context: acc[dt] += P(16x32) @ V(32x16), V pre-transposed [d][s]
    #pragma unroll
    for (int dt = 0; dt < 4; dt++) {
      FragB16 bv;
      const unsigned short* vrow =
          Vbase + (size_t)(dt * 16 + lr) * SS + k0 + kh16;
      bv.q[0] = *(const uint4*)(vrow);
      bv.q[1] = *(const uint4*)(vrow + 8);
      acc[dt].v = __builtin_amdgcn_wmma_f32_16x16x32_bf16(
          false, ap.v, false, bv.v, (short)0, acc[dt].v, false, false);
    }
  }

  float invl[8];
  #pragma unroll
  for (int r = 0; r < 8; r++) invl[r] = 1.0f / lrun[r];

  // store context bf16 as [B,S,D]
  #pragma unroll
  for (int dt = 0; dt < 4; dt++)
    #pragma unroll
    for (int r = 0; r < 8; r++) {
      int row = r + rowhalf;
      ctx[((size_t)(b * SS) + q0 + row) * DD + h * DH + dt * 16 + lr] =
          f2bf(acc[dt].f[r] * invl[r]);
    }

  // -------------------- pass 2: write attn weights --------------------
  float* arow = attn + ((size_t)bh * SS + q0) * SS;
  for (int kt = 0; kt < 64; kt++) {
    const int k0 = kt * 32;
    FragB16 bk[2][2];
    #pragma unroll
    for (int kc = 0; kc < 2; kc++) {
      const unsigned short* krow = Kbase + (size_t)(k0 + kc * 16 + lr) * DH;
      #pragma unroll
      for (int ds = 0; ds < 2; ds++) {
        bk[kc][ds].q[0] = *(const uint4*)(krow + ds * 32 + kh16);
        bk[kc][ds].q[1] = *(const uint4*)(krow + ds * 32 + kh16 + 8);
      }
    }
    FragF32 c0, c1;
    c0.v = __builtin_amdgcn_wmma_f32_16x16x32_bf16(false, aq0.v, false, bk[0][0].v,
                                                   (short)0, zero8, false, false);
    c0.v = __builtin_amdgcn_wmma_f32_16x16x32_bf16(false, aq1.v, false, bk[0][1].v,
                                                   (short)0, c0.v, false, false);
    c1.v = __builtin_amdgcn_wmma_f32_16x16x32_bf16(false, aq0.v, false, bk[1][0].v,
                                                   (short)0, zero8, false, false);
    c1.v = __builtin_amdgcn_wmma_f32_16x16x32_bf16(false, aq1.v, false, bk[1][1].v,
                                                   (short)0, c1.v, false, false);

    const float mv0 = mk[k0 + lr] * -1.0e9f;
    const float mv1 = mk[k0 + 16 + lr] * -1.0e9f;
    #pragma unroll
    for (int r = 0; r < 8; r++) {
      int row = r + rowhalf;
      arow[(size_t)row * SS + k0 + lr] =
          __expf(c0.f[r] * scale + mv0 - mrun[r]) * invl[r];
      arow[(size_t)row * SS + k0 + 16 + lr] =
          __expf(c1.f[r] * scale + mv1 - mrun[r]) * invl[r];
    }
  }
}

// ---------------------------------------------------------------------------
// Kernel 3: output projection.  out[4096,1024] = ctx(bf16) @ Wo + bo, fp32.
// A tile staged with global_load_async_to_lds_b128 (ASYNCcnt), overlapping
// the fp32->bf16 W-tile conversion on the vector pipe.
// ---------------------------------------------------------------------------
__global__ __launch_bounds__(256) void out_proj_kernel(
    const unsigned short* __restrict__ ctx, const float* __restrict__ Wo,
    const float* __restrict__ bo, float* __restrict__ out)
{
  __shared__ unsigned short ldsA[64][32];
  __shared__ unsigned short ldsW[64][32];

  const int tid  = threadIdx.x;
  const int lane = tid & 31;
  const int warp = tid >> 5;
  const int m0 = blockIdx.y * 64;
  const int n0 = blockIdx.x * 64;
  const int m0w = (warp >> 1) * 16;
  const int n0w = (warp & 1) * 32;
  const int lr   = lane & 15;
  const int koff = (lane >> 4) * 8;
  const int kh16 = (lane >> 4) * 16;

  FragF32 c0, c1;
  #pragma unroll
  for (int i = 0; i < 8; i++) { c0.f[i] = 0.f; c1.f[i] = 0.f; }

  const int arow_ = tid >> 2, acol = (tid & 3) * 8;
  const unsigned lds_off =
      (unsigned)(uintptr_t)&ldsA[arow_][acol];           // LDS byte offset

  for (int kk = 0; kk < DD; kk += 32) {
    { // stage A (already bf16): async global -> LDS, 16B per lane
      unsigned long long gaddr = (unsigned long long)(uintptr_t)(
          ctx + (size_t)(m0 + arow_) * DD + kk + acol);
      asm volatile("global_load_async_to_lds_b128 %0, %1, off"
                   :: "v"(lds_off), "v"(gaddr) : "memory");
    }
    { // stage Wo fp32 -> bf16 transposed to [n][k]
      int row = tid >> 3, col = (tid & 7) * 8;
      const float* src = Wo + (size_t)(kk + row) * DD + n0 + col;
      #pragma unroll
      for (int j = 0; j < 8; j++) ldsW[col + j][row] = f2bf(src[j]);
    }
    asm volatile("s_wait_asynccnt 0x0" ::: "memory");
    __syncthreads();

    FragB16 a, b0, b1;
    a.q[0]  = *(const uint4*)&ldsA[m0w + lr][kh16];
    a.q[1]  = *(const uint4*)&ldsA[m0w + lr][kh16 + 8];
    b0.q[0] = *(const uint4*)&ldsW[n0w + lr][kh16];
    b0.q[1] = *(const uint4*)&ldsW[n0w + lr][kh16 + 8];
    b1.q[0] = *(const uint4*)&ldsW[n0w + 16 + lr][kh16];
    b1.q[1] = *(const uint4*)&ldsW[n0w + 16 + lr][kh16 + 8];

    c0.v = __builtin_amdgcn_wmma_f32_16x16x32_bf16(false, a.v, false, b0.v,
                                                   (short)0, c0.v, false, false);
    c1.v = __builtin_amdgcn_wmma_f32_16x16x32_bf16(false, a.v, false, b1.v,
                                                   (short)0, c1.v, false, false);
    __syncthreads();
  }

  const int rowhalf = (lane >> 4) * 8;
  #pragma unroll
  for (int cn = 0; cn < 2; cn++) {
    FragF32& c = cn ? c1 : c0;
    int gn = n0 + n0w + cn * 16 + lr;
    float bia = bo[gn];
    #pragma unroll
    for (int r = 0; r < 8; r++) {
      int gm = m0 + m0w + rowhalf + r;
      out[(size_t)gm * DD + gn] = c.f[r] + bia;
    }
  }
}

// ---------------------------------------------------------------------------
extern "C" void kernel_launch(void* const* d_in, const int* in_sizes, int n_in,
                              void* d_out, int out_size, void* d_ws, size_t ws_size,
                              hipStream_t stream) {
  (void)in_sizes; (void)n_in; (void)out_size; (void)ws_size;
  const float* query = (const float*)d_in[0];
  const float* key_  = (const float*)d_in[1];
  const float* value = (const float*)d_in[2];
  const float* mask  = (const float*)d_in[3];
  const float* Wq = (const float*)d_in[4];  const float* bq = (const float*)d_in[5];
  const float* Wk = (const float*)d_in[6];  const float* bk = (const float*)d_in[7];
  const float* Wv = (const float*)d_in[8];  const float* bv = (const float*)d_in[9];
  const float* Wo = (const float*)d_in[10]; const float* bo = (const float*)d_in[11];

  float* out  = (float*)d_out;                       // [B,S,D]
  float* attn = out + (size_t)BB * SS * DD;          // [B,H,S,S]

  // workspace: Qp | Kp | Vt | ctx, each B*S*D bf16 = 8 MiB (32 MiB total)
  const size_t n_elems = (size_t)BB * SS * DD;
  unsigned short* Qp  = (unsigned short*)d_ws;
  unsigned short* Kp  = Qp + n_elems;
  unsigned short* Vt  = Kp + n_elems;
  unsigned short* ctx = Vt + n_elems;

  dim3 gProj(DD / 64, (BB * SS) / 64, 3);
  qkv_proj_kernel<<<gProj, 256, 0, stream>>>(query, key_, value,
                                             Wq, bq, Wk, bk, Wv, bv,
                                             Qp, Kp, Vt);

  // one wave per (b,h,q-tile16): 2*16*128 = 4096 waves, 4 waves/block
  attn_kernel<<<dim3(1024), 128, 0, stream>>>(Qp, Kp, Vt, mask, attn, ctx);

  dim3 gOut(DD / 64, (BB * SS) / 64);
  out_proj_kernel<<<gOut, 256, 0, stream>>>(ctx, Wo, bo, out);
}
